// MultiHeadSelfAttention_51024211476739
// MI455X (gfx1250) — matrix operand order repaired and checked by reading, not compiled, not run on hardware
//
#include <hip/hip_runtime.h>

typedef _Float16 half_t;
typedef __attribute__((ext_vector_type(16))) _Float16 v16h;
typedef __attribute__((ext_vector_type(8)))  _Float16 v8h;
typedef __attribute__((ext_vector_type(8)))  float    v8f;

static constexpr int kC  = 512;     // embed dim
static constexpr int kS  = 4096;    // H*W
static constexpr int kB  = 16;      // batch

#if __has_builtin(__builtin_amdgcn_global_load_async_to_lds_b128)
#define ASYNC_LDS 1
// parameter type leaked by the round-2 diagnostic: int4 __device__ *
typedef __attribute__((vector_size(16))) int i128;
typedef __attribute__((address_space(1))) i128 gi128;   // global
typedef __attribute__((address_space(3))) i128 li128;   // LDS
#endif

__device__ __forceinline__ v8f wmma16x16x32(v16h a, v16h b, v8f c) {
  return __builtin_amdgcn_wmma_f32_16x16x32_f16(false, a, false, b, (short)0, c,
                                                false, false);
}

__device__ __forceinline__ v16h frag_cat(v8h lo, v8h hi) {
  return __builtin_shufflevector(lo, hi, 0, 1, 2, 3, 4, 5, 6, 7, 8, 9, 10, 11,
                                 12, 13, 14, 15);
}

// Fragment-swizzle helpers (A: 16x32 f16, B: 32x16 f16, contiguous k order
// k = 16*(lane/16) + j).  Storage unit: [tile][lane(32)][j(16)] -> 512 halves/tile.
__device__ __forceinline__ int a_lane(int m, int kk) {   // m in [0,16), kk in [0,32)
  return ((kk >> 4) << 4) + m;
}
__device__ __forceinline__ int a_elem(int kk) {
  return kk & 15;
}
__device__ __forceinline__ int b_lane(int n, int kk) {   // n in [0,16)
  return ((kk >> 4) << 4) + n;
}
__device__ __forceinline__ int b_elem(int kk) { return kk & 15; }

// ---------------------------------------------------------------------------
// Projection GEMM: p[o,s] = sum_c W[o,c] * x[b,c,s] + bias[o], scattered into
// WMMA-fragment-swizzled global layouts:
// MODE 0: q -> qA [bh][kt(16)][mb(4)] tiles      (A operand of scores GEMM)
// MODE 1: k -> kB [bh][kt(16)][nt(4)] tiles      (B operand of scores GEMM)
// MODE 2: v -> vB [bh][chunk(64)][ks(2)][nt(4)]  (B operand of attend GEMM)
// grid: (S/64, M/64, B), block: 128 threads (4 waves)
// ---------------------------------------------------------------------------
template <int MODE>
__global__ __launch_bounds__(128)
void proj_kernel(const float* __restrict__ x, const float* __restrict__ W,
                 const float* __restrict__ bias, half_t* __restrict__ out) {
  __shared__ __align__(32) half_t As[64][40];   // 64 rows x 32 K, row layout
  __shared__ __align__(32) half_t Bsw[2048];    // 32x64 tile, frag-swizzled

  const int tid  = threadIdx.x;
  const int lane = tid & 31;
  const int wv   = tid >> 5;
  const int b    = blockIdx.z;
  const int m0   = blockIdx.y << 6;
  const int n0   = blockIdx.x << 6;

  const float* xb = x + (size_t)b * kC * kS;

  const v8f vzero = {0.f, 0.f, 0.f, 0.f, 0.f, 0.f, 0.f, 0.f};
  v8f acc[4];
#pragma unroll
  for (int i = 0; i < 4; ++i) acc[i] = vzero;

  const int ar = tid >> 1, ac = (tid & 1) << 4;   // A stage: row, col-base
  const int br = tid >> 2, bnt = tid & 3;         // B stage: k-row, n-subtile

  for (int k0 = 0; k0 < kC; k0 += 32) {
    // ---- stage A tile: W[m0+ar][k0+ac..+16), f32 -> f16, row layout ----
    {
      const float* src = W + (size_t)(m0 + ar) * kC + k0 + ac;
      float4 f0 = ((const float4*)src)[0];
      float4 f1 = ((const float4*)src)[1];
      float4 f2 = ((const float4*)src)[2];
      float4 f3 = ((const float4*)src)[3];
      half_t* d = &As[ar][ac];
      d[0]  = (half_t)f0.x; d[1]  = (half_t)f0.y; d[2]  = (half_t)f0.z; d[3]  = (half_t)f0.w;
      d[4]  = (half_t)f1.x; d[5]  = (half_t)f1.y; d[6]  = (half_t)f1.z; d[7]  = (half_t)f1.w;
      d[8]  = (half_t)f2.x; d[9]  = (half_t)f2.y; d[10] = (half_t)f2.z; d[11] = (half_t)f2.w;
      d[12] = (half_t)f3.x; d[13] = (half_t)f3.y; d[14] = (half_t)f3.z; d[15] = (half_t)f3.w;
    }
    // ---- stage B tile swizzled: x[k0+br][n0+bnt*16..+16) ----
    {
      const float* src = xb + (size_t)(k0 + br) * kS + n0 + (bnt << 4);
      if (k0 + 32 < kC) __builtin_prefetch(src + (size_t)32 * kS, 0, 1);
      float4 f0 = ((const float4*)src)[0];
      float4 f1 = ((const float4*)src)[1];
      float4 f2 = ((const float4*)src)[2];
      float4 f3 = ((const float4*)src)[3];
      // element (kk=br, n=i): offset = (bnt*32 + (br>>4)*16 + i)*16 + (br&15)
      half_t* d = Bsw + (bnt << 9) + ((br >> 4) << 8) + (br & 15);
      d[0*16]  = (half_t)f0.x; d[1*16]  = (half_t)f0.y; d[2*16]  = (half_t)f0.z; d[3*16]  = (half_t)f0.w;
      d[4*16]  = (half_t)f1.x; d[5*16]  = (half_t)f1.y; d[6*16]  = (half_t)f1.z; d[7*16]  = (half_t)f1.w;
      d[8*16]  = (half_t)f2.x; d[9*16]  = (half_t)f2.y; d[10*16] = (half_t)f2.z; d[11*16] = (half_t)f2.w;
      d[12*16] = (half_t)f3.x; d[13*16] = (half_t)f3.y; d[14*16] = (half_t)f3.z; d[15*16] = (half_t)f3.w;
    }
    __syncthreads();

    const half_t* ap = &As[(wv << 4) + (lane & 15)][0];
    const int kb = (lane >> 4) << 4;
    v16h af = frag_cat(*(const v8h*)(ap + kb), *(const v8h*)(ap + kb + 8));
#pragma unroll
    for (int nt = 0; nt < 4; ++nt) {
      v16h bf = *(const v16h*)(Bsw + (nt << 9) + (lane << 4));
      acc[nt] = wmma16x16x32(af, bf, acc[nt]);
    }
    __syncthreads();
  }

  // Epilogue: bias + fused head-permute scatter into swizzled layouts
  const int mrel = (wv << 4) + ((lane >> 4) << 3);
  const int nrel = lane & 15;
#pragma unroll
  for (int nt = 0; nt < 4; ++nt) {
#pragma unroll
    for (int i = 0; i < 8; ++i) {
      const int o = m0 + mrel + i;
      const int s = n0 + (nt << 4) + nrel;
      const float val = acc[nt][i] + bias[o];
      size_t addr;
      if (MODE == 0) {        // q: d=s&63, l=(o&7)*64+s/64, h=o>>3
        const int bh = b * 8 + (o >> 3);
        const int d = s & 63, l = ((o & 7) << 6) + (s >> 6);
        const int kt = l >> 5, kk = l & 31;
        addr = ((size_t)bh << 15) + ((size_t)((kt << 2) + (d >> 4)) << 9) +
               (a_lane(d & 15, kk) << 4) + a_elem(kk);
      } else if (MODE == 1) { // k: e=s&63, l as above
        const int bh = b * 8 + (o >> 3);
        const int e = s & 63, l = ((o & 7) << 6) + (s >> 6);
        const int kt = l >> 5, kk = l & 31;
        addr = ((size_t)bh << 15) + ((size_t)((kt << 2) + (e >> 4)) << 9) +
               (b_lane(e & 15, kk) << 4) + b_elem(kk);
      } else {                // v: h=o>>6, e=s&63, n=(o&63)*64+s/64
        const int bh = b * 8 + (o >> 6);
        const int e = s & 63, n = ((o & 63) << 6) + (s >> 6);
        const int kk = e & 31;
        const int tile = (((n >> 6) << 1) + (e >> 5)) * 4 + ((n >> 4) & 3);
        addr = ((size_t)bh << 18) + ((size_t)tile << 9) +
               (b_lane(n & 15, kk) << 4) + b_elem(kk);
      }
      out[addr] = (half_t)val;
    }
  }
}

// ---------------------------------------------------------------------------
// Attention core per (b,h): scores = q*k^T/8 (64x64, K=512) with BOTH operands
// read as coalesced pre-swizzled fragments from global (no LDS, no barriers),
// softmax in LDS, attend GEMM with A from swizzled LDS + B fragments straight
// from global.  grid: (128 bh, 8 n-splits), block: 128 threads
// ---------------------------------------------------------------------------
__global__ __launch_bounds__(128)
void attn_kernel(const half_t* __restrict__ qAg, const half_t* __restrict__ kBg,
                 const half_t* __restrict__ vBg, half_t* __restrict__ obuf) {
  __shared__ float  sc[64][68];
  __shared__ __align__(32) half_t attnSw[4096];   // 64x64 attn, frag-swizzled

  const int tid = threadIdx.x, lane = tid & 31, wv = tid >> 5;
  const int bh = blockIdx.x, b = bh >> 3, h = bh & 7;

  const half_t* qA = qAg + ((size_t)bh << 15);
  const half_t* kB = kBg + ((size_t)bh << 15);
  const half_t* vB = vBg + ((size_t)bh << 18);

  const v8f vzero = {0.f, 0.f, 0.f, 0.f, 0.f, 0.f, 0.f, 0.f};
  v8f acc[4];
#pragma unroll
  for (int i = 0; i < 4; ++i) acc[i] = vzero;

  // ---- scores GEMM: pure global_load_b128 + wmma stream ----
  for (int kt = 0; kt < 16; ++kt) {
    v16h af = *(const v16h*)(qA + (((kt << 2) + wv) << 9) + (lane << 4));
#pragma unroll
    for (int nt = 0; nt < 4; ++nt) {
      v16h bf = *(const v16h*)(kB + (((kt << 2) + nt) << 9) + (lane << 4));
      acc[nt] = wmma16x16x32(af, bf, acc[nt]);
    }
  }

  const int mrel = (wv << 4) + ((lane >> 4) << 3);
  const int nrel = lane & 15;
#pragma unroll
  for (int nt = 0; nt < 4; ++nt)
#pragma unroll
    for (int i = 0; i < 8; ++i)
      sc[mrel + i][(nt << 4) + nrel] = acc[nt][i] * 0.125f;   // 1/sqrt(64)
  __syncthreads();

  // ---- softmax: one row per thread; write attn swizzled for A-fragments ----
  if (tid < 64) {
    float mx = -3.4e38f;
    for (int j = 0; j < 64; ++j) mx = fmaxf(mx, sc[tid][j]);
    float sum = 0.f;
    for (int j = 0; j < 64; ++j) {
      float e = __expf(sc[tid][j] - mx);
      sc[tid][j] = e;
      sum += e;
    }
    const float inv = 1.f / sum;
    const int m = tid & 15, mb = tid >> 4;
    for (int e = 0; e < 64; ++e) {
      const int kk = e & 31;
      attnSw[((((e >> 5) << 2) + mb) << 9) + (a_lane(m, kk) << 4) + a_elem(kk)] =
          (half_t)(sc[tid][e] * inv);
    }
  }
  __syncthreads();

  // ---- attend GEMM: attn(64x64) x v(64x4096) in 64-col chunks ----
  for (int ci = 0; ci < 8; ++ci) {
    const int chunk = (blockIdx.y << 3) + ci;
    v8f oc[4];
#pragma unroll
    for (int i = 0; i < 4; ++i) oc[i] = vzero;

#pragma unroll
    for (int ks = 0; ks < 2; ++ks) {
      v16h af = *(const v16h*)(attnSw + (((ks << 2) + wv) << 9) + (lane << 4));
      const half_t* vt = vB + ((size_t)(((chunk << 1) + ks) << 2) << 9);
#pragma unroll
      for (int nt = 0; nt < 4; ++nt) {
        v16h bf = *(const v16h*)(vt + (nt << 9) + (lane << 4));
        oc[nt] = wmma16x16x32(af, bf, oc[nt]);
      }
    }

    // scatter (packed v8h, contiguous along d) into the flat buffer that the
    // final projection reads as the torch [HW, C] reshape view
    const size_t base = ((size_t)(b * 512 + chunk * 8 + h)) << 12;
#pragma unroll
    for (int nt = 0; nt < 4; ++nt) {
      v8h pk;
#pragma unroll
      for (int i = 0; i < 8; ++i) pk[i] = (half_t)oc[nt][i];
      const int ncol = (nt << 4) + nrel;
      *(v8h*)(obuf + base + (ncol << 6) + mrel) = pk;
    }
  }
}

// ---------------------------------------------------------------------------
// Final projection: y[b,s,o] = sum_c view[b,s,c] * Wo[o,c] + bo[o]
// grid: (512/64, 4096/64, B), block: 128 threads
// ---------------------------------------------------------------------------
__global__ __launch_bounds__(128)
void outproj_kernel(const half_t* __restrict__ abuf, const float* __restrict__ Wo,
                    const float* __restrict__ bo, float* __restrict__ y) {
  __shared__ __align__(32) half_t As[64][40];
  __shared__ __align__(32) half_t Bsw[2048];

  const int tid = threadIdx.x, lane = tid & 31, wv = tid >> 5;
  const int b  = blockIdx.z;
  const int m0 = blockIdx.y << 6;   // s
  const int n0 = blockIdx.x << 6;   // o

  const half_t* Ab = abuf + (size_t)b * kS * kC;

  const v8f vzero = {0.f, 0.f, 0.f, 0.f, 0.f, 0.f, 0.f, 0.f};
  v8f acc[4];
#pragma unroll
  for (int i = 0; i < 4; ++i) acc[i] = vzero;

  const int ar = tid >> 1, ac = (tid & 1) << 4;

  for (int k0 = 0; k0 < kC; k0 += 32) {
    // ---- stage A tile (f16 copy; async CDNA5 path when available) ----
    {
      const half_t* src = Ab + (size_t)(m0 + ar) * kC + k0 + ac;
#if defined(ASYNC_LDS)
      __builtin_amdgcn_global_load_async_to_lds_b128(
          (gi128*)src, (li128*)&As[ar][ac], 0, 0);
      __builtin_amdgcn_global_load_async_to_lds_b128(
          (gi128*)(src + 8), (li128*)&As[ar][ac + 8], 0, 0);
#else
      v8h* d = (v8h*)&As[ar][ac];
      d[0] = ((const v8h*)src)[0];
      d[1] = ((const v8h*)src)[1];
#endif
    }
    // ---- stage B tile swizzled + transposed: Bsw <- Wo[o][c] ----
    {
      const int orow = tid >> 1;           // n (0..63)
      const int occ  = (tid & 1) << 4;     // kk base (0 or 16)
      const float* src = Wo + (size_t)(n0 + orow) * kC + k0 + occ;
      float4 f0 = ((const float4*)src)[0];
      float4 f1 = ((const float4*)src)[1];
      float4 f2 = ((const float4*)src)[2];
      float4 f3 = ((const float4*)src)[3];
      // (nt = orow>>4, lane = (occ>>4)*16 + (orow&15), j = kk&15) contiguous
      half_t* d = Bsw + ((orow >> 4) << 9) + ((occ >> 4) << 8) + ((orow & 15) << 4);
      d[0]  = (half_t)f0.x; d[1]  = (half_t)f0.y; d[2]  = (half_t)f0.z; d[3]  = (half_t)f0.w;
      d[4]  = (half_t)f1.x; d[5]  = (half_t)f1.y; d[6]  = (half_t)f1.z; d[7]  = (half_t)f1.w;
      d[8]  = (half_t)f2.x; d[9]  = (half_t)f2.y; d[10] = (half_t)f2.z; d[11] = (half_t)f2.w;
      d[12] = (half_t)f3.x; d[13] = (half_t)f3.y; d[14] = (half_t)f3.z; d[15] = (half_t)f3.w;
    }
#if defined(ASYNC_LDS)
#if __has_builtin(__builtin_amdgcn_s_wait_asynccnt)
    __builtin_amdgcn_s_wait_asynccnt(0);
#else
    asm volatile("s_wait_asynccnt 0" ::: "memory");
#endif
#endif
    __syncthreads();

    const half_t* ap = &As[(wv << 4) + (lane & 15)][0];
    const int kb = (lane >> 4) << 4;
    v16h af = frag_cat(*(const v8h*)(ap + kb), *(const v8h*)(ap + kb + 8));
#pragma unroll
    for (int nt = 0; nt < 4; ++nt) {
      v16h bf = *(const v16h*)(Bsw + (nt << 9) + (lane << 4));
      acc[nt] = wmma16x16x32(af, bf, acc[nt]);
    }
    __syncthreads();
  }

  const int mrel = (wv << 4) + ((lane >> 4) << 3);
  const int nrel = lane & 15;
#pragma unroll
  for (int nt = 0; nt < 4; ++nt) {
#pragma unroll
    for (int i = 0; i < 8; ++i) {
      const int s = m0 + mrel + i;
      const int o = n0 + (nt << 4) + nrel;
      y[((size_t)b * kS + s) * kC + o] = acc[nt][i] + bo[o];
    }
  }
}

// ---------------------------------------------------------------------------
extern "C" void kernel_launch(void* const* d_in, const int* in_sizes, int n_in,
                              void* d_out, int out_size, void* d_ws, size_t ws_size,
                              hipStream_t stream) {
  const float* x  = (const float*)d_in[0];
  const float* Wq = (const float*)d_in[1];
  const float* bq = (const float*)d_in[2];
  const float* Wk = (const float*)d_in[3];
  const float* bk = (const float*)d_in[4];
  const float* Wv = (const float*)d_in[5];
  const float* bv = (const float*)d_in[6];
  const float* Wo = (const float*)d_in[7];
  const float* bo = (const float*)d_in[8];
  float* out = (float*)d_out;

  // workspace carve-up (all f16, fragment-swizzled):
  half_t* qA   = (half_t*)d_ws;                            // 128 * 32768
  half_t* kBw  = qA  + ((size_t)kB * 8 << 15);
  half_t* vBw  = kBw + ((size_t)kB * 8 << 15);             // 128 * 262144
  half_t* abuf = vBw + ((size_t)kB * 8 << 18);             // 16*512*4096
  (void)in_sizes; (void)n_in; (void)out_size; (void)ws_size;

  dim3 blk(128);
  proj_kernel<0><<<dim3(kS / 64, 1, kB), blk, 0, stream>>>(x, Wq, bq, qA);
  proj_kernel<1><<<dim3(kS / 64, 1, kB), blk, 0, stream>>>(x, Wk, bk, kBw);
  proj_kernel<2><<<dim3(kS / 64, kC / 64, kB), blk, 0, stream>>>(x, Wv, bv, vBw);
  attn_kernel<<<dim3(kB * 8, 8), blk, 0, stream>>>(qA, kBw, vBw, abuf);
  outproj_kernel<<<dim3(kC / 64, kS / 64, kB), blk, 0, stream>>>(abuf, Wo, bo, out);
}
